// ReBasedCausalSelfAttention_21242908246346
// MI455X (gfx1250) — compile-verified, matrix operand order checked by reading, not run on hardware
//
#include <hip/hip_runtime.h>

// ---------------------------------------------------------------------------
// ReBased causal linear attention for MI455X (gfx1250, wave32, WMMA)
// B=2, T=2048, C=1024, H=16, D=64
// Pipeline: f32->bf16 convert -> bf16 WMMA QKV GEMM (LDS-staged, double
// buffered, 32x64 wave tile) -> f32 LN+scan -> bf16 WMMA proj GEMM
// ---------------------------------------------------------------------------

#define BB 2
#define TT 2048
#define CC 1024
#define HH 16
#define DD 64
#define EPSF 1e-6f

// LDS W-tile row pitch in elements (32 data + 8 pad) to spread LDS banks
#define WPITCH 40

typedef __attribute__((ext_vector_type(16))) __bf16 v16bf;
typedef __attribute__((ext_vector_type(8)))  float  v8f;

union FragU { uint4 u4[2]; v16bf v; };

__device__ __forceinline__ unsigned short f2bf_bits(float f) {
    unsigned int u = __builtin_bit_cast(unsigned int, f);
    u += 0x7FFFu + ((u >> 16) & 1u);            // round to nearest even
    return (unsigned short)(u >> 16);
}
__device__ __forceinline__ float bf2f(unsigned short b) {
    unsigned int u = ((unsigned int)b) << 16;
    return __builtin_bit_cast(float, u);
}

// ---------------------------------------------------------------------------
// f32 -> bf16 conversion (bandwidth trivial; runs once per buffer)
// ---------------------------------------------------------------------------
__global__ __launch_bounds__(256) void f32_to_bf16_kernel(const float* __restrict__ in,
                                                          unsigned short* __restrict__ out,
                                                          int n) {
    int i = blockIdx.x * blockDim.x + threadIdx.x;
    if (i < n) out[i] = f2bf_bits(in[i]);
}

// ---------------------------------------------------------------------------
// C[M,N] (bf16 or f32) = A[M,K] @ W[N,K]^T, bf16 inputs, f32 accumulate.
// Block = 256 threads = 8 waves; block tile = 256(M) x 64(N), K-step 32.
// Wave tile = 32(M) x 64(N): 2 A fragments x 4 B fragments -> 8 WMMAs per
// K-step against the same 8 ds_load_b128 of B (AI 2x vs 16x64 tile).
// W tile (64x32 bf16, 4KB) staged once per block per K-step through LDS,
// double buffered: global loads of step k+1 issue before the WMMAs of step k,
// ds_store lands after them, one barrier per step.
// Fragment layout per CDNA5 ISA 7.12.2 (16-bit A/B):
//   lanes 0-15:  row=lane,    K in {k0..k0+7, k0+16..k0+23}
//   lanes 16-31: row=lane-16, K in {k0+8..k0+15, k0+24..k0+31}
// ---------------------------------------------------------------------------
template <bool OUTF32>
__global__ __launch_bounds__(256) void gemm_bf16_wmma(const unsigned short* __restrict__ A,
                                                      const unsigned short* __restrict__ W,
                                                      unsigned short* __restrict__ Cbf,
                                                      float* __restrict__ Cf,
                                                      int M, int N, int K) {
    __shared__ __align__(16) unsigned short sW[2][64 * WPITCH];

    const int tid   = threadIdx.x;
    const int lane  = tid & 31;
    const int wave  = tid >> 5;
    const int Mb    = blockIdx.y * 256 + wave * 32;
    const int Nb    = blockIdx.x * 64;
    const int lrow  = lane & 15;
    const int khalf = (lane >> 4) << 3;          // 0 or 8

    const unsigned short* Ap0 = A + (size_t)(Mb + lrow) * K + khalf;
    const unsigned short* Ap1 = A + (size_t)(Mb + 16 + lrow) * K + khalf;

    // staging map: 256 threads x 16B cover the 64x32 tile (8 elems each)
    const int srow = tid >> 2;                   // 0..63
    const int scol = (tid & 3) * 8;              // 0,8,16,24
    const unsigned short* Wg = W + (size_t)(Nb + srow) * K + scol;
    unsigned short* sWst0 = &sW[0][srow * WPITCH + scol];
    unsigned short* sWst1 = &sW[1][srow * WPITCH + scol];

    v8f acc[8] = {};

    // ---- prologue: A frags k=0 + stage W tile k=0 ----
    FragU fa[2];
    fa[0].u4[0] = *(const uint4*)(Ap0);
    fa[0].u4[1] = *(const uint4*)(Ap0 + 16);
    fa[1].u4[0] = *(const uint4*)(Ap1);
    fa[1].u4[1] = *(const uint4*)(Ap1 + 16);
    *(uint4*)sWst0 = *(const uint4*)(Wg);
    __syncthreads();

    const int NK = K >> 5;
    for (int kk = 0; kk < NK; ++kk) {
        const int  buf  = kk & 1;
        const bool more = (kk + 1 < NK);

        // ---- issue next-step global loads first (hide behind WMMAs) ----
        FragU fan[2];
        uint4 wst;
        if (more) {
            const int kn = (kk + 1) << 5;
            __builtin_prefetch(Ap0 + kn + 128, 0, 3);
            wst           = *(const uint4*)(Wg + kn);
            fan[0].u4[0]  = *(const uint4*)(Ap0 + kn);
            fan[0].u4[1]  = *(const uint4*)(Ap0 + kn + 16);
            fan[1].u4[0]  = *(const uint4*)(Ap1 + kn);
            fan[1].u4[1]  = *(const uint4*)(Ap1 + kn + 16);
        }

        // ---- pull B fragments for 4 N-tiles from LDS (all live at once) ----
        FragU fb[4];
#pragma unroll
        for (int t = 0; t < 4; ++t) {
            const unsigned short* sp = &sW[buf][(t * 16 + lrow) * WPITCH + khalf];
            fb[t].u4[0] = *(const uint4*)(sp);
            fb[t].u4[1] = *(const uint4*)(sp + 16);
        }

        // ---- 8 WMMAs: each B fragment reused by both A fragments ----
#pragma unroll
        for (int t = 0; t < 4; ++t) {
#pragma unroll
            for (int j = 0; j < 2; ++j) {
                acc[t * 2 + j] = __builtin_amdgcn_wmma_f32_16x16x32_bf16(
                    false, fa[j].v, false, fb[t].v, (short)0, acc[t * 2 + j], false, false);
            }
        }

        // ---- land the staged tile + rotate A buffers ----
        if (more) {
            unsigned short* dst = (buf ? sWst0 : sWst1);
            *(uint4*)dst = wst;
            fa[0] = fan[0];
            fa[1] = fan[1];
        }
        __syncthreads();
    }

    // D layout: acc[t*2+j][i] -> row Mb + j*16 + i + 8*(lane>>4),
    //                           col Nb + t*16 + (lane&15)
    const int mbase = Mb + ((lane >> 4) << 3);
    const int ncol  = Nb + lrow;
#pragma unroll
    for (int t = 0; t < 4; ++t) {
#pragma unroll
        for (int j = 0; j < 2; ++j) {
#pragma unroll
            for (int i = 0; i < 8; ++i) {
                size_t idx = (size_t)(mbase + j * 16 + i) * N + (ncol + t * 16);
                if (OUTF32) Cf[idx]  = acc[t * 2 + j][i];
                else        Cbf[idx] = f2bf_bits(acc[t * 2 + j][i]);
            }
        }
    }
}

// ---------------------------------------------------------------------------
// Per-(b,h) causal linear-attention scan with register-resident 64x64 state.
// Block = 256 threads: thread j owns S[d0..d0+15][e], e = j&63, d0 = (j>>6)*16.
// Wave 0 computes LN(q), LN(k), phi=(g*xn+b)^2, running Z and den via shfl_xor.
// ---------------------------------------------------------------------------
__global__ __launch_bounds__(256) void rebased_scan_kernel(const unsigned short* __restrict__ qkv,
                                                           const float* __restrict__ gamma,
                                                           const float* __restrict__ beta,
                                                           unsigned short* __restrict__ y) {
    const int bh = blockIdx.x;
    const int b  = bh >> 4;
    const int h  = bh & 15;

    __shared__ float s_phq[DD], s_phk[DD], s_v[DD];
    __shared__ float s_np[4][DD];
    __shared__ float s_gam[DD], s_bet[DD];
    __shared__ float s_den;

    const int tid  = threadIdx.x;
    const int lane = tid & 31;
    const int e    = tid & 63;
    const int dg   = tid >> 6;
    const int d0   = dg * 16;

    if (tid < DD) { s_gam[tid] = gamma[h * DD + tid]; s_bet[tid] = beta[h * DD + tid]; }

    float S[16];
#pragma unroll
    for (int i = 0; i < 16; ++i) S[i] = 0.f;
    float z0 = 0.f, z1 = 0.f;    // running Z (wave 0 lanes only)

    __syncthreads();

    for (int t = 0; t < TT; ++t) {
        if (tid < 32) {
            const size_t row = (size_t)(b * TT + t) * (3 * CC);
            const int c0 = h * DD + lane;
            float q0 = bf2f(qkv[row + c0]);
            float q1 = bf2f(qkv[row + c0 + 32]);
            float k0 = bf2f(qkv[row + CC + c0]);
            float k1 = bf2f(qkv[row + CC + c0 + 32]);
            float v0 = bf2f(qkv[row + 2 * CC + c0]);
            float v1 = bf2f(qkv[row + 2 * CC + c0 + 32]);

            // --- LN(q) ---
            float sm = q0 + q1;
#pragma unroll
            for (int m = 16; m >= 1; m >>= 1) sm += __shfl_xor(sm, m, 32);
            float mean = sm * (1.f / 64.f);
            float a0 = q0 - mean, a1 = q1 - mean;
            float ss = a0 * a0 + a1 * a1;
#pragma unroll
            for (int m = 16; m >= 1; m >>= 1) ss += __shfl_xor(ss, m, 32);
            float inv = rsqrtf(ss * (1.f / 64.f) + EPSF);
            float hq0 = s_gam[lane] * a0 * inv + s_bet[lane];
            float hq1 = s_gam[lane + 32] * a1 * inv + s_bet[lane + 32];
            float pq0 = hq0 * hq0, pq1 = hq1 * hq1;
            s_phq[lane] = pq0; s_phq[lane + 32] = pq1;

            // --- LN(k) ---
            sm = k0 + k1;
#pragma unroll
            for (int m = 16; m >= 1; m >>= 1) sm += __shfl_xor(sm, m, 32);
            mean = sm * (1.f / 64.f);
            a0 = k0 - mean; a1 = k1 - mean;
            ss = a0 * a0 + a1 * a1;
#pragma unroll
            for (int m = 16; m >= 1; m >>= 1) ss += __shfl_xor(ss, m, 32);
            inv = rsqrtf(ss * (1.f / 64.f) + EPSF);
            float hk0 = s_gam[lane] * a0 * inv + s_bet[lane];
            float hk1 = s_gam[lane + 32] * a1 * inv + s_bet[lane + 32];
            float pk0 = hk0 * hk0, pk1 = hk1 * hk1;
            s_phk[lane] = pk0; s_phk[lane + 32] = pk1;

            z0 += pk0; z1 += pk1;                 // inclusive cumsum of phi_k
            s_v[lane] = v0; s_v[lane + 32] = v1;

            float dp = pq0 * z0 + pq1 * z1;       // den = sum(phi_q * Z)
#pragma unroll
            for (int m = 16; m >= 1; m >>= 1) dp += __shfl_xor(dp, m, 32);
            if (lane == 0) s_den = dp + EPSF;
        }
        __syncthreads();

        // rank-1 state update + contraction (inclusive: update S before use)
        {
            float ve = s_v[e];
            float np = 0.f;
#pragma unroll
            for (int i = 0; i < 16; ++i) {
                float pk = s_phk[d0 + i];
                S[i] = fmaf(pk, ve, S[i]);
                np   = fmaf(s_phq[d0 + i], S[i], np);
            }
            s_np[dg][e] = np;
        }
        __syncthreads();

        if (tid < DD) {
            float num = s_np[0][tid] + s_np[1][tid] + s_np[2][tid] + s_np[3][tid];
            float yv  = num / s_den;
            y[(size_t)(b * TT + t) * CC + h * DD + tid] = f2bf_bits(yv);
        }
        __syncthreads();
    }
}

// ---------------------------------------------------------------------------
// Launch
// ---------------------------------------------------------------------------
extern "C" void kernel_launch(void* const* d_in, const int* in_sizes, int n_in,
                              void* d_out, int out_size, void* d_ws, size_t ws_size,
                              hipStream_t stream) {
    (void)in_sizes; (void)n_in; (void)out_size; (void)ws_size;
    const float* x     = (const float*)d_in[0];
    const float* Wqkv  = (const float*)d_in[1];
    const float* Wproj = (const float*)d_in[2];
    const float* gamma = (const float*)d_in[3];
    const float* beta  = (const float*)d_in[4];
    float* out = (float*)d_out;

    const size_t nx   = (size_t)BB * TT * CC;      // 4.19M
    const size_t nwq  = (size_t)3 * CC * CC;       // 3.15M
    const size_t nwp  = (size_t)CC * CC;           // 1.05M
    const size_t nqkv = (size_t)BB * TT * 3 * CC;  // 12.6M

    unsigned short* x_bf     = (unsigned short*)d_ws;
    unsigned short* wqkv_bf  = x_bf + nx;
    unsigned short* wproj_bf = wqkv_bf + nwq;
    unsigned short* qkv_bf   = wproj_bf + nwp;
    unsigned short* y_bf     = qkv_bf + nqkv;      // total ~48 MiB of ws

    // 1) convert inputs to bf16
    f32_to_bf16_kernel<<<(int)((nx  + 255) / 256), 256, 0, stream>>>(x,     x_bf,    (int)nx);
    f32_to_bf16_kernel<<<(int)((nwq + 255) / 256), 256, 0, stream>>>(Wqkv,  wqkv_bf, (int)nwq);
    f32_to_bf16_kernel<<<(int)((nwp + 255) / 256), 256, 0, stream>>>(Wproj, wproj_bf,(int)nwp);

    // 2) qkv = x @ Wqkv^T   (M=4096, N=3072, K=1024)
    {
        dim3 grid(3 * CC / 64, BB * TT / 256);
        gemm_bf16_wmma<false><<<grid, 256, 0, stream>>>(x_bf, wqkv_bf, qkv_bf, nullptr,
                                                        BB * TT, 3 * CC, CC);
    }

    // 3) phi + causal linear-attention scan (32 independent (b,h) scans)
    rebased_scan_kernel<<<BB * HH, 256, 0, stream>>>(qkv_bf, gamma, beta, y_bf);

    // 4) out = y @ Wproj^T  (M=4096, N=1024, K=1024), f32 output
    {
        dim3 grid(CC / 64, BB * TT / 256);
        gemm_bf16_wmma<true><<<grid, 256, 0, stream>>>(y_bf, wproj_bf, nullptr, out,
                                                       BB * TT, CC, CC);
    }
}
